// AdvancedIntegratedFiberOpticsNN_9036611190931
// MI455X (gfx1250) — matrix operand order repaired and checked by reading, not compiled, not run on hardware
//
#include <hip/hip_runtime.h>
#include <hip/hip_bf16.h>

// ---------------------------------------------------------------------------
// MI455X (gfx1250) implementation.
// Heavy stages (pattern 3x3 conv K=864, cls1 1x1 K=192, cls2 1x1 K=256) run as
// f16 WMMA GEMMs (v_wmma_f32_16x16x32_f16, f32 accumulate) over pixel-major
// f16 tensors so all A/B operand fetches are contiguous 16B b128 loads with
// compile-time IOFFSETs. Stage-1 (Cin=3 convs), Sobel reduction and the
// 5-wide heads stay VALU f32 (~5% of FLOPs).
// ---------------------------------------------------------------------------

typedef __attribute__((ext_vector_type(16))) _Float16 v16h;
typedef __attribute__((ext_vector_type(8)))  _Float16 v8h;
typedef __attribute__((ext_vector_type(8)))  float    v8f;

#define HH 256
#define WW 256
#define BB 4
#define HWC 65536              // H*W
#define CXT 192                // channels/pixel in combined_t (96 feat + 48 sim + 24 am + 24 pad)
#define KPAT 864               // 9 taps * 96 ci

// workspace layout (bytes)
#define OFF_RED   0                       // float[8]: avg_g_sum[4], avg_p_sum, pad
#define OFF_WPAT  256                     // f16 [96][864]
#define OFF_WC1   (OFF_WPAT + 96*864*2)   // f16 [256][192]
#define OFF_WC2   (OFF_WC1 + 256*192*2)   // f16 [128][256]
#define OFF_XT    (OFF_WC2 + 128*256*2)   // f16 [B*HW][192]  (~96 MB)

static __device__ inline v8f zf() {
  v8f r;
#pragma unroll
  for (int i = 0; i < 8; ++i) r[i] = 0.f;
  return r;
}
static __device__ inline v8h zh() {
  v8h r;
#pragma unroll
  for (int i = 0; i < 8; ++i) r[i] = (_Float16)0.f;
  return r;
}
static __device__ inline v16h mk16(v8h lo, v8h hi) {
  v16h r;
#pragma unroll
  for (int i = 0; i < 8; ++i) { r[i] = lo[i]; r[i + 8] = hi[i]; }
  return r;
}
static __device__ inline v8f wmma_f16(v8h alo, v8h ahi, v8h blo, v8h bhi, v8f c) {
  return __builtin_amdgcn_wmma_f32_16x16x32_f16(
      false, mk16(alo, ahi), false, mk16(blo, bhi), (short)0, c, false, false);
}
static __device__ inline float sigmoidf(float v) {
  return 1.f / (1.f + __expf(-v));
}

// ---------------------------------------------------------------------------
__global__ __launch_bounds__(32) void k_init(float* red) {
  if (threadIdx.x < 8) red[threadIdx.x] = 0.f;
}

// pattern weights -> f16 [96 rows][K=864], K index = tap*96 + ci, rows 72..95 zero
__global__ __launch_bounds__(256) void k_pack_pat(const float* __restrict__ np,
                                                  const float* __restrict__ apw,
                                                  _Float16* __restrict__ wpat) {
  int i = blockIdx.x * 256 + threadIdx.x;
  if (i >= 96 * KPAT) return;
  int co = i / KPAT, k = i % KPAT;
  int tap = k / 96, ci = k % 96;
  float v = 0.f;
  if (co < 48)      v = np[co * 864 + ci * 9 + tap];
  else if (co < 72) v = apw[(co - 48) * 864 + ci * 9 + tap];
  wpat[i] = (_Float16)v;
}

__global__ __launch_bounds__(256) void k_pack_c1(const float* __restrict__ w1,
                                                 _Float16* __restrict__ wc1) {
  int i = blockIdx.x * 256 + threadIdx.x;
  if (i >= 256 * CXT) return;
  int co = i / CXT, k = i % CXT;
  wc1[i] = (_Float16)(k < 168 ? w1[co * 168 + k] : 0.f);
}

__global__ __launch_bounds__(256) void k_pack_c2(const float* __restrict__ w2,
                                                 _Float16* __restrict__ wc2) {
  int i = blockIdx.x * 256 + threadIdx.x;
  if (i >= 128 * 256) return;
  wc2[i] = (_Float16)w2[i];
}

// ---------------------------------------------------------------------------
// Sobel-magnitude mean per batch + radial-position mean (batch 0 only).
__global__ __launch_bounds__(256) void k_reduce(const float* __restrict__ x,
                                                float* __restrict__ red) {
  int tid = threadIdx.x;
  int flat = blockIdx.x * 256 + tid;
  int b = flat >> 16;
  int y = (flat >> 8) & 255;
  int xc = flat & 255;
  size_t base = ((size_t)b * 3) << 16;

  auto gray = [&](int yy, int xx) -> float {
    if (yy < 0 || yy > 255 || xx < 0 || xx > 255) return 0.f;
    size_t o = (size_t)yy * 256 + xx;
    return (x[base + o] + x[base + HWC + o] + x[base + 2 * HWC + o]) * (1.f / 3.f);
  };
  float gmm = gray(y - 1, xc - 1), gm0 = gray(y - 1, xc), gmp = gray(y - 1, xc + 1);
  float g0m = gray(y, xc - 1),                           g0p = gray(y, xc + 1);
  float gpm = gray(y + 1, xc - 1), gp0 = gray(y + 1, xc), gpp = gray(y + 1, xc + 1);
  float gx = (gmp + 2.f * g0p + gpp) - (gmm + 2.f * g0m + gpm);
  float gy = (gpm + 2.f * gp0 + gpp) - (gmm + 2.f * gm0 + gmp);
  float mag = sqrtf(gx * gx + gy * gy);

  __shared__ float s[256];
  s[tid] = mag;
  __syncthreads();
#pragma unroll
  for (int st = 128; st > 0; st >>= 1) {
    if (tid < st) s[tid] += s[tid + st];
    __syncthreads();
  }
  if (tid == 0) atomicAdd(&red[b], s[0]);

  if (b == 0) {
    __syncthreads();
    float nx = -1.f + 2.f * (float)xc / 255.f;
    float ny = -1.f + 2.f * (float)y / 255.f;
    s[tid] = sqrtf(nx * nx + ny * ny);
    __syncthreads();
#pragma unroll
    for (int st = 128; st > 0; st >>= 1) {
      if (tid < st) s[tid] += s[tid + st];
      __syncthreads();
    }
    if (tid == 0) atomicAdd(&red[4], s[0]);
  }
}

// ---------------------------------------------------------------------------
// Stage-1 convs (Cin=3, 3x3/5x5/7x7) -> f16 combined_t channels 0..95.
template <int KS, int P>
static __device__ void branch_conv(const float* xs, int ty, int tx,
                                   const float* __restrict__ w,
                                   const float* __restrict__ bias,
                                   float scale, _Float16* op) {
#pragma unroll
  for (int cg = 0; cg < 4; ++cg) {
    v8h o;
#pragma unroll
    for (int r = 0; r < 8; ++r) {
      int co = cg * 8 + r;
      float a = bias[co];
      for (int ci = 0; ci < 3; ++ci)
#pragma unroll
        for (int ky = 0; ky < KS; ++ky)
#pragma unroll
          for (int kx = 0; kx < KS; ++kx)
            a += xs[(ci * 22 + (ty + 3 + ky - P)) * 22 + (tx + 3 + kx - P)] *
                 w[((co * 3 + ci) * KS + ky) * KS + kx];
      o[r] = (_Float16)(a * scale);
    }
    *(v8h*)(op + cg * 8) = o;
  }
}

__global__ __launch_bounds__(256) void k_feat(
    const float* __restrict__ x,
    const float* __restrict__ w3, const float* __restrict__ b3,
    const float* __restrict__ w5, const float* __restrict__ b5,
    const float* __restrict__ w7, const float* __restrict__ b7,
    const float* __restrict__ gw, const float* __restrict__ pw,
    const float* __restrict__ gadj, const float* __restrict__ padj,
    const float* __restrict__ red, _Float16* __restrict__ xt) {
  __shared__ float xs[3 * 22 * 22];
  int b = blockIdx.x >> 8;
  int rem = blockIdx.x & 255;
  int y0 = (rem >> 4) << 4;
  int x0 = (rem & 15) << 4;
  int tid = threadIdx.x;

  for (int i = tid; i < 3 * 22 * 22; i += 256) {
    int c = i / 484, r2 = i % 484, yy = r2 / 22, xx = r2 % 22;
    int gy = y0 + yy - 3, gx0 = x0 + xx - 3;
    float v = 0.f;
    if (gy >= 0 && gy < 256 && gx0 >= 0 && gx0 < 256)
      v = x[(((size_t)b * 3 + c) << 16) + (size_t)gy * 256 + gx0];
    xs[i] = v;
  }
  __syncthreads();

  int ty = tid >> 4, tx = tid & 15;
  float avg_g = red[b] * (1.f / (float)HWC);
  float avg_p = red[4] * (1.f / (float)HWC);
  float ga = gadj[0], pa = padj[0];
  float s0 = (1.f + gw[0] * avg_g * ga) * (1.f + pw[0] * avg_p * pa);
  float s1 = (1.f + gw[1] * avg_g * ga) * (1.f + pw[1] * avg_p * pa);
  float s2 = (1.f + gw[2] * avg_g * ga) * (1.f + pw[2] * avg_p * pa);

  size_t pix = (size_t)b * HWC + (size_t)(y0 + ty) * 256 + (x0 + tx);
  _Float16* op = xt + pix * CXT;
  branch_conv<3, 1>(xs, ty, tx, w3, b3, s0, op);
  branch_conv<5, 2>(xs, ty, tx, w5, b5, s1, op + 32);
  branch_conv<7, 3>(xs, ty, tx, w7, b7, s2, op + 64);
}

// ---------------------------------------------------------------------------
// Pattern 3x3 conv as WMMA GEMM: out[72+pad, px] = Wpat[96][864] * im2col(feat).
// One wave = 16 co x 16 px tile; 9 taps x 3 ci-blocks = 27 v_wmma per wave.
// Interior tiles take a branch-free path: one per-lane base pointer, every
// tap/cb/half offset is a compile-time constant (folds into load IOFFSET).
__global__ __launch_bounds__(256) void k_pattern(_Float16* __restrict__ xt,
                                                 const _Float16* __restrict__ wpat) {
  int wid = (blockIdx.x * blockDim.x + threadIdx.x) >> 5;
  int lane = threadIdx.x & 31;
  int hf = lane >> 4;      // half-wave select
  int lrow = lane & 15;    // A row (M) / B column (N)
  int cot = wid % 6;       // co tile 0..5 -> channels 96..191 of combined_t
  int pt = wid / 6;        // pixel tile
  int px0 = pt * 16;
  int b = px0 >> 16;
  int y = (px0 >> 8) & 255;
  int x0 = px0 & 255;

  const _Float16* arow = wpat + (size_t)(cot * 16 + lrow) * KPAT + 8 * hf;
  const _Float16* bbase =
      xt + ((size_t)b * HWC + (size_t)y * 256 + (x0 + lrow)) * CXT + 8 * hf;
  v8f acc = zf();

  bool interior = (y >= 1) & (y <= 254) & (x0 >= 16) & (x0 <= 224);  // uniform
  if (interior) {
#pragma unroll
    for (int ky = 0; ky < 3; ++ky)
#pragma unroll
      for (int kx = 0; kx < 3; ++kx)
#pragma unroll
        for (int cb = 0; cb < 3; ++cb) {
          const _Float16* apt = arow + (ky * 3 + kx) * 96 + cb * 32;
          v8h alo = *(const v8h*)(apt);
          v8h ahi = *(const v8h*)(apt + 16);
          const _Float16* bpt =
              bbase + (ky - 1) * (256 * CXT) + (kx - 1) * CXT + cb * 32;
          v8h blo = *(const v8h*)(bpt);
          v8h bhi = *(const v8h*)(bpt + 16);
          acc = wmma_f16(alo, ahi, blo, bhi, acc);
        }
  } else {
#pragma unroll
    for (int ky = 0; ky < 3; ++ky) {
      int yy = y + ky - 1;
      if (yy < 0 || yy > 255) continue;        // zero-pad row contributes nothing
#pragma unroll
      for (int kx = 0; kx < 3; ++kx) {
        int xx = x0 + lrow + kx - 1;
        bool vx = (xx >= 0 && xx <= 255);
#pragma unroll
        for (int cb = 0; cb < 3; ++cb) {
          const _Float16* apt = arow + (ky * 3 + kx) * 96 + cb * 32;
          v8h alo = *(const v8h*)(apt);
          v8h ahi = *(const v8h*)(apt + 16);
          v8h blo = zh(), bhi = zh();
          if (vx) {
            const _Float16* bpt =
                bbase + (ky - 1) * (256 * CXT) + (kx - 1) * CXT + cb * 32;
            blo = *(const v8h*)(bpt);
            bhi = *(const v8h*)(bpt + 16);
          }
          acc = wmma_f16(alo, ahi, blo, bhi, acc);
        }
      }
    }
  }

  // D: element r -> channel co = cot*16 + 8*hf + r, column px = px0 + lrow
  size_t pix = (size_t)b * HWC + (size_t)y * 256 + x0 + lrow;
  int cbase = cot * 16 + 8 * hf;
  v8h outv;
#pragma unroll
  for (int r = 0; r < 8; ++r) {
    int co = cbase + r;
    float v = (co < 72) ? sigmoidf(acc[r]) : 0.f;   // sim(48)+am(24), pad->0
    outv[r] = (_Float16)v;
  }
  *(v8h*)(xt + pix * CXT + 96 + cbase) = outv;
}

// ---------------------------------------------------------------------------
// Fused cls1 (192->256) -> cls2 (256->128) -> 5 heads, one 16-pixel tile/block.
__global__ __launch_bounds__(512) void k_cls(
    const _Float16* __restrict__ xt,
    const _Float16* __restrict__ wc1, const float* __restrict__ b1,
    const _Float16* __restrict__ wc2, const float* __restrict__ b2,
    const float* __restrict__ rw, const float* __restrict__ rb,
    const float* __restrict__ aw, const float* __restrict__ ab,
    const float* __restrict__ qw, const float* __restrict__ qb,
    const float* __restrict__ thr, float* __restrict__ out) {
  __shared__ v8h hbuf[16 * 32];   // h  [px][256] f16
  __shared__ v8h pbuf[16 * 16];   // p  [px][128] f16
  int tid = threadIdx.x;
  int wave = tid >> 5;
  int lane = tid & 31;
  int hf = lane >> 4;
  int lrow = lane & 15;
  int px0 = blockIdx.x * 16;

  // ---- phase 1: h = relu(Wc1 * combined + b1), 16 waves x 16 co each
  {
    int cobase = wave * 16;
    const _Float16* apt0 = wc1 + (size_t)(cobase + lrow) * CXT + 8 * hf;
    const _Float16* bpt0 = xt + (size_t)(px0 + lrow) * CXT + 8 * hf;
    v8f acc = zf();
#pragma unroll
    for (int cb = 0; cb < 6; ++cb) {
      v8h alo = *(const v8h*)(apt0 + cb * 32);
      v8h ahi = *(const v8h*)(apt0 + cb * 32 + 16);
      v8h blo = *(const v8h*)(bpt0 + cb * 32);
      v8h bhi = *(const v8h*)(bpt0 + cb * 32 + 16);
      acc = wmma_f16(alo, ahi, blo, bhi, acc);
    }
    int c8 = cobase + 8 * hf;
    v8h hv;
#pragma unroll
    for (int r = 0; r < 8; ++r) {
      float v = acc[r] + b1[c8 + r];
      hv[r] = (_Float16)(v > 0.f ? v : 0.f);
    }
    hbuf[lrow * 32 + (c8 >> 3)] = hv;
  }
  __syncthreads();

  // ---- phase 2: processed = relu(Wc2 * h + b2), 8 waves x 16 co each
  if (wave < 8) {
    int cobase = wave * 16;
    const _Float16* apt0 = wc2 + (size_t)(cobase + lrow) * 256 + 8 * hf;
    v8f acc = zf();
#pragma unroll
    for (int cb = 0; cb < 8; ++cb) {
      v8h alo = *(const v8h*)(apt0 + cb * 32);
      v8h ahi = *(const v8h*)(apt0 + cb * 32 + 16);
      v8h blo = hbuf[lrow * 32 + cb * 4 + hf];
      v8h bhi = hbuf[lrow * 32 + cb * 4 + 2 + hf];
      acc = wmma_f16(alo, ahi, blo, bhi, acc);
    }
    int c8 = cobase + 8 * hf;
    v8h pv;
#pragma unroll
    for (int r = 0; r < 8; ++r) {
      float v = acc[r] + b2[c8 + r];
      pv[r] = (_Float16)(v > 0.f ? v : 0.f);
    }
    pbuf[lrow * 16 + (c8 >> 3)] = pv;
  }
  __syncthreads();

  // ---- phase 3: 5-wide heads + anomaly-norm mean (one lane per pixel).
  // processed read as 16x ds_load_b128; all weight fetches are lane-uniform.
  if (tid < 16) {
    size_t pix = (size_t)px0 + tid;
    int b = (int)(pix >> 16);
    int y = (int)((pix >> 8) & 255);
    int xc = (int)(pix & 255);
    float d0 = rb[0], d1 = rb[1], d2 = rb[2];
    float sa = ab[0], sq = qb[0];
    for (int kk = 0; kk < 16; ++kk) {
      v8h pv8 = pbuf[tid * 16 + kk];
#pragma unroll
      for (int r = 0; r < 8; ++r) {
        float p = (float)pv8[r];
        int k = kk * 8 + r;
        d0 += p * rw[k];
        d1 += p * rw[128 + k];
        d2 += p * rw[256 + k];
        sa += p * aw[k];
        sq += p * qw[k];
      }
    }
    const v8h* simv = (const v8h*)(xt + pix * CXT + 96);
    float an = 0.f;
    for (int jj = 0; jj < 6; ++jj) {
      v8h sv = simv[jj];
#pragma unroll
      for (int r = 0; r < 8; ++r) {
        float t = thr[jj * 8 + r] - (float)sv[r];
        an += (t > 0.f ? t : 0.f);
      }
    }
    float d3 = sigmoidf(sa) + an * (1.f / 48.f);
    float d4 = sigmoidf(sq);
    size_t obase = ((size_t)b * 5) * HWC + (size_t)y * 256 + xc;
    out[obase]           = d0;
    out[obase + HWC]     = d1;
    out[obase + 2 * HWC] = d2;
    out[obase + 3 * HWC] = d3;
    out[obase + 4 * HWC] = d4;
  }
}

// ---------------------------------------------------------------------------
extern "C" void kernel_launch(void* const* d_in, const int* in_sizes, int n_in,
                              void* d_out, int out_size, void* d_ws, size_t ws_size,
                              hipStream_t stream) {
  (void)in_sizes; (void)n_in; (void)out_size; (void)ws_size;
  const float* x    = (const float*)d_in[0];
  const float* w3   = (const float*)d_in[1];
  const float* b3   = (const float*)d_in[2];
  const float* w5   = (const float*)d_in[3];
  const float* b5   = (const float*)d_in[4];
  const float* w7   = (const float*)d_in[5];
  const float* b7   = (const float*)d_in[6];
  const float* gw   = (const float*)d_in[7];
  const float* pw   = (const float*)d_in[8];
  const float* gadj = (const float*)d_in[9];
  const float* padj = (const float*)d_in[10];
  const float* np   = (const float*)d_in[11];
  const float* thr  = (const float*)d_in[12];
  const float* apw  = (const float*)d_in[13];
  const float* w1   = (const float*)d_in[14];
  const float* b1   = (const float*)d_in[15];
  const float* w2   = (const float*)d_in[16];
  const float* b2   = (const float*)d_in[17];
  const float* rw   = (const float*)d_in[18];
  const float* rb   = (const float*)d_in[19];
  const float* aw   = (const float*)d_in[20];
  const float* ab   = (const float*)d_in[21];
  const float* qw   = (const float*)d_in[22];
  const float* qb   = (const float*)d_in[23];

  char* ws = (char*)d_ws;
  float*     red  = (float*)(ws + OFF_RED);
  _Float16*  wpat = (_Float16*)(ws + OFF_WPAT);
  _Float16*  wc1  = (_Float16*)(ws + OFF_WC1);
  _Float16*  wc2  = (_Float16*)(ws + OFF_WC2);
  _Float16*  xt   = (_Float16*)(ws + OFF_XT);

  k_init<<<1, 32, 0, stream>>>(red);
  k_pack_pat<<<(96 * KPAT + 255) / 256, 256, 0, stream>>>(np, apw, wpat);
  k_pack_c1<<<(256 * CXT + 255) / 256, 256, 0, stream>>>(w1, wc1);
  k_pack_c2<<<(128 * 256 + 255) / 256, 256, 0, stream>>>(w2, wc2);
  k_reduce<<<(BB * HWC) / 256, 256, 0, stream>>>(x, red);
  k_feat<<<BB * 256, 256, 0, stream>>>(x, w3, b3, w5, b5, w7, b7,
                                       gw, pw, gadj, padj, red, xt);
  // 16384 pixel tiles * 6 co tiles = 98304 waves, 8 waves/block
  k_pattern<<<(16384 * 6) / 8, 256, 0, stream>>>(xt, wpat);
  k_cls<<<16384, 512, 0, stream>>>(xt, wc1, b1, wc2, b2,
                                   rw, rb, aw, ab, qw, qb, thr, (float*)d_out);
}
